// M5_BN_brevitas_44418551775869
// MI455X (gfx1250) — 8-bit matrix operand order repaired and checked by reading, compiled, not run on hardware
//
#include <hip/hip_runtime.h>
#include <hip/hip_bf16.h>
#include <cstdint>

typedef __attribute__((ext_vector_type(8))) int v8i;

#define WAVES_PER_BLOCK 4
#define NT 4

__global__ void k_sign_i8(const float* __restrict__ in, int8_t* __restrict__ out, int n) {
    int i = blockIdx.x * blockDim.x + threadIdx.x;
    if (i < n) out[i] = (in[i] >= 0.0f) ? (int8_t)1 : (int8_t)-1;
}

__global__ void k_wprep(const float* __restrict__ w, int8_t* __restrict__ wb,
                        int Cout, int Cin, int KS, int Kpad) {
    int i = blockIdx.x * blockDim.x + threadIdx.x;
    if (i >= Cout * Kpad) return;
    int cout = i / Kpad, t = i - cout * Kpad;
    int Ktot = Cin * KS;
    int8_t v = 0;
    if (t < Ktot) {
        int ks = t / Cin, cin = t - ks * Cin;
        v = (w[((long)cout * Cin + cin) * KS + ks] >= 0.0f) ? (int8_t)1 : (int8_t)-1;
    }
    wb[i] = v;
}

__global__ void k_zero_halo(int8_t* __restrict__ buf, int Bn, int Lpitch, int C, int halo) {
    int i = blockIdx.x * blockDim.x + threadIdx.x;
    int per_b = 2 * halo * C;
    if (i >= Bn * per_b) return;
    int b = i / per_b, r = i - b * per_b;
    int j = r / C, c = r - j * C;
    int posidx = (j < halo) ? j : (Lpitch - 2 * halo + j);
    buf[((long)b * Lpitch + posidx) * C + c] = 0;
}

__global__ void k_bnprep(const float* __restrict__ g, const float* __restrict__ b,
                         const float* __restrict__ m, const float* __restrict__ v,
                         float* __restrict__ inv, float* __restrict__ beta, int n) {
    int i = blockIdx.x * blockDim.x + threadIdx.x;
    if (i >= n) return;
    float iv = g[i] * rsqrtf(v[i] + 1e-5f);
    inv[i]  = iv;
    beta[i] = b[i] - m[i] * iv;
}

template <int CL>
__global__ __launch_bounds__(32 * WAVES_PER_BLOCK)
void k_bconv(const int8_t* __restrict__ in, const int8_t* __restrict__ wt,
             int8_t* __restrict__ out8, float* __restrict__ outf,
             const float* __restrict__ bninv, const float* __restrict__ bnbeta,
             int N, int Kpad, int Lip, int Cout, int Lout, int Lop,
             int stride, int pad) {
    const int lane = threadIdx.x & 31;
    const int wave = threadIdx.x >> 5;
    const int half = lane >> 4;
    const int l16  = lane & 15;
    const int mbase = blockIdx.x * 16;
    const int nbase = (blockIdx.y * WAVES_PER_BLOCK + wave) * (16 * NT);

    const int8_t* arow = wt + (long)(mbase + l16) * Kpad;

    bool nval[NT];
    long obase[NT];
    const int8_t* bcol[NT];
#pragma unroll
    for (int t = 0; t < NT; ++t) {
        int n   = nbase + t * 16 + l16;
        nval[t] = (n < N);
        int nc  = nval[t] ? n : (N - 1);
        int bb  = nc / Lout;
        int ll  = nc - bb * Lout;
        int pos0 = ll * stride - pad;
        bcol[t]  = in + ((((long)bb * Lip) + pos0) << CL);
        obase[t] = ((long)bb * Lop + ll) * Cout;
    }

    v8i acc[NT] = {};

    for (int kc = 0; kc < Kpad; kc += 64) {
        v8i a;
#pragma unroll
        for (int g = 0; g < 4; ++g) {
            uint2 wv = *(const uint2*)(arow + kc + ((g >> 1) << 5) + (half << 4) + ((g & 1) << 3));
            a[2 * g]     = (int)wv.x;
            a[2 * g + 1] = (int)wv.y;
        }
#pragma unroll
        for (int t = 0; t < NT; ++t) {
            v8i bf;
            if constexpr (CL == 0) {
#pragma unroll
                for (int v = 0; v < 8; ++v) {
                    int kb = kc + ((v >> 2) << 5) + (half << 4) + ((v & 3) << 2);
                    bf[v] = (int)*(const unsigned*)(bcol[t] + kb);
                }
            } else {
                uint4 lo = *(const uint4*)(bcol[t] + kc + (half << 4));
                uint4 hi = *(const uint4*)(bcol[t] + kc + 32 + (half << 4));
                bf[0] = (int)lo.x; bf[1] = (int)lo.y; bf[2] = (int)lo.z; bf[3] = (int)lo.w;
                bf[4] = (int)hi.x; bf[5] = (int)hi.y; bf[6] = (int)hi.z; bf[7] = (int)hi.w;
            }
            acc[t] = __builtin_amdgcn_wmma_i32_16x16x64_iu8(true, a, true, bf,
                                                            acc[t], false, false);
        }
    }

#pragma unroll
    for (int t = 0; t < NT; ++t) {
        if (!nval[t]) continue;
        if (outf) {
#pragma unroll
            for (int r = 0; r < 8; ++r)
                outf[obase[t] + mbase + (half << 3) + r] = (float)acc[t][r];
        } else {
            unsigned lo = 0u, hi = 0u;
#pragma unroll
            for (int r = 0; r < 8; ++r) {
                int   m = mbase + (half << 3) + r;
                float x = (float)acc[t][r];
                float y = bninv ? fmaf(x, bninv[m], bnbeta[m]) : x;
                unsigned byt = (y >= 0.0f) ? 0x01u : 0xFFu;
                if (r < 4) lo |= byt << (8 * r);
                else       hi |= byt << (8 * (r - 4));
            }
            *(uint2*)(out8 + obase[t] + mbase + (half << 3)) = make_uint2(lo, hi);
        }
    }
}

__global__ void k_fc1(const float* __restrict__ in, const float* __restrict__ w,
                      const float* __restrict__ inv, const float* __restrict__ beta,
                      int8_t* __restrict__ out, int B, int C) {
    int idx = blockIdx.x * blockDim.x + threadIdx.x;
    if (idx >= B * C) return;
    int b = idx / C, j = idx % C;
    float s = 0.0f;
    for (int c = 0; c < C; ++c) {
        float wv = (w[j * C + c] >= 0.0f) ? 1.0f : -1.0f;
        s = fmaf(in[b * C + c], wv, s);
    }
    float y = fmaf(s, inv[j], beta[j]);
    out[idx] = (y >= 0.0f) ? (int8_t)1 : (int8_t)-1;
}

__global__ void k_fc2(const int8_t* __restrict__ in, const float* __restrict__ w,
                      float* __restrict__ out, int B, int C, int NC) {
    int idx = blockIdx.x * blockDim.x + threadIdx.x;
    if (idx >= B * NC) return;
    int b = idx / NC, j = idx % NC;
    float s = 0.0f;
    for (int c = 0; c < C; ++c) {
        float wv = (w[j * C + c] >= 0.0f) ? 1.0f : -1.0f;
        s = fmaf((float)in[b * C + c], wv, s);
    }
    out[idx] = s;
}

template <int CL>
static void run_conv(const int8_t* in, const int8_t* wt,
                     int8_t* out8, float* outf,
                     const float* bninv, const float* bnbeta,
                     int B, int Kpad, int Lip, int Cout, int Lout, int Lop,
                     int stride, int pad, hipStream_t s) {
    const int cols_per_block = 16 * NT * WAVES_PER_BLOCK;
    dim3 grid(Cout / 16, (B * Lout + cols_per_block - 1) / cols_per_block);
    k_bconv<CL><<<grid, 32 * WAVES_PER_BLOCK, 0, s>>>(
        in, wt, out8, outf, bninv, bnbeta,
        B * Lout, Kpad, Lip, Cout, Lout, Lop, stride, pad);
}

static inline void run_sign(const float* in, int8_t* out, int n, hipStream_t s) {
    k_sign_i8<<<(n + 255) / 256, 256, 0, s>>>(in, out, n);
}

static inline void run_wprep(const float* w, int8_t* wb, int Cout, int Cin, int KS,
                             int Kpad, hipStream_t s) {
    int n = Cout * Kpad;
    k_wprep<<<(n + 255) / 256, 256, 0, s>>>(w, wb, Cout, Cin, KS, Kpad);
}

extern "C" void kernel_launch(void* const* d_in, const int* in_sizes, int n_in,
                              void* d_out, int out_size, void* d_ws, size_t ws_size,
                              hipStream_t stream) {
    (void)in_sizes; (void)n_in; (void)out_size; (void)ws_size;

    constexpr int B = 64, C = 128, C2 = 256, NC = 35;
    constexpr int L0 = 16000, L1 = 3980, L2 = 995, L3 = 996, L4 = 249,
                  L5 = 248, L6 = 62, L7 = 64, L8 = 16;
    constexpr int HALO = 2;
    constexpr int LP2 = L2 + 2 * HALO, LP4 = L4 + 2 * HALO, LP6 = L6 + 2 * HALO;

    const float* x    = (const float*)d_in[0];
    const float* w1   = (const float*)d_in[1];
    const float* p1w  = (const float*)d_in[2];
    const float* w2   = (const float*)d_in[3];
    const float* p2w  = (const float*)d_in[4];
    const float* w3   = (const float*)d_in[5];
    const float* p3w  = (const float*)d_in[6];
    const float* w5   = (const float*)d_in[7];
    const float* p5w  = (const float*)d_in[8];
    const float* embw = (const float*)d_in[9];
    const float* fc1w = (const float*)d_in[10];
    const float* fc2w = (const float*)d_in[11];
    const float* bn1p[4] = {(const float*)d_in[12], (const float*)d_in[13],
                            (const float*)d_in[14], (const float*)d_in[15]};
    const float* bn2p[4] = {(const float*)d_in[16], (const float*)d_in[17],
                            (const float*)d_in[18], (const float*)d_in[19]};
    const float* bn3p[4] = {(const float*)d_in[20], (const float*)d_in[21],
                            (const float*)d_in[22], (const float*)d_in[23]};
    const float* bn5p[4] = {(const float*)d_in[24], (const float*)d_in[25],
                            (const float*)d_in[26], (const float*)d_in[27]};
    const float* bnfp[4] = {(const float*)d_in[28], (const float*)d_in[29],
                            (const float*)d_in[30], (const float*)d_in[31]};
    float* out = (float*)d_out;

    char*  base = (char*)d_ws;
    size_t off  = 0;
    auto alloc = [&](size_t bytes) -> char* {
        char* p = base + off;
        off = (off + bytes + 255) & ~(size_t)255;
        return p;
    };
    int8_t* xb    = (int8_t*)alloc((size_t)B * L0 + 256);
    int8_t* w1b   = (int8_t*)alloc((size_t)C * 128);
    int8_t* p1wb  = (int8_t*)alloc((size_t)C * C * 4);
    int8_t* w2b   = (int8_t*)alloc((size_t)C * C * 4);
    int8_t* p2wb  = (int8_t*)alloc((size_t)C * C * 4);
    int8_t* w3b   = (int8_t*)alloc((size_t)C2 * C * 4);
    int8_t* p3wb  = (int8_t*)alloc((size_t)C2 * C2 * 4);
    int8_t* w5b   = (int8_t*)alloc((size_t)C2 * C2 * 3);
    int8_t* p5wb  = (int8_t*)alloc((size_t)C2 * C2 * 4);
    int8_t* embwb = (int8_t*)alloc((size_t)C2 * C2 * 16);
    float* bn1i = (float*)alloc(C * 4);   float* bn1b = (float*)alloc(C * 4);
    float* bn2i = (float*)alloc(C * 4);   float* bn2b = (float*)alloc(C * 4);
    float* bn3i = (float*)alloc(C2 * 4);  float* bn3b = (float*)alloc(C2 * 4);
    float* bn5i = (float*)alloc(C2 * 4);  float* bn5b = (float*)alloc(C2 * 4);
    float* bnfi = (float*)alloc(C2 * 4);  float* bnfb = (float*)alloc(C2 * 4);
    int8_t* h1  = (int8_t*)alloc((size_t)B * L1 * C);
    int8_t* h2p = (int8_t*)alloc((size_t)B * LP2 * C);
    int8_t* h4p = (int8_t*)alloc((size_t)B * LP4 * C);
    int8_t* h6p = (int8_t*)alloc((size_t)B * LP6 * C2);
    int8_t* h3  = (int8_t*)alloc((size_t)B * L3 * C);
    int8_t* h5  = (int8_t*)alloc((size_t)B * L5 * C2);
    int8_t* h7  = (int8_t*)alloc((size_t)B * L7 * C2);
    int8_t* h8  = (int8_t*)alloc((size_t)B * L8 * C2);
    float*  h9f = (float*)alloc((size_t)B * C2 * 4);
    int8_t* h10 = (int8_t*)alloc((size_t)B * C2);
    int8_t* h2 = h2p + (size_t)HALO * C;
    int8_t* h4 = h4p + (size_t)HALO * C;
    int8_t* h6 = h6p + (size_t)HALO * C2;

    run_sign(x, xb, B * L0, stream);
    run_wprep(w1,   w1b,   C,  1,  84, 128,  stream);
    run_wprep(p1w,  p1wb,  C,  C,  4,  512,  stream);
    run_wprep(w2,   w2b,   C,  C,  4,  512,  stream);
    run_wprep(p2w,  p2wb,  C,  C,  4,  512,  stream);
    run_wprep(w3,   w3b,   C2, C,  4,  512,  stream);
    run_wprep(p3w,  p3wb,  C2, C2, 4,  1024, stream);
    run_wprep(w5,   w5b,   C2, C2, 3,  768,  stream);
    run_wprep(p5w,  p5wb,  C2, C2, 4,  1024, stream);
    run_wprep(embw, embwb, C2, C2, 16, 4096, stream);

    {
        int n2 = B * 2 * HALO * C;
        k_zero_halo<<<(n2 + 255) / 256, 256, 0, stream>>>(h2p, B, LP2, C, HALO);
        int n4 = B * 2 * HALO * C;
        k_zero_halo<<<(n4 + 255) / 256, 256, 0, stream>>>(h4p, B, LP4, C, HALO);
        int n6 = B * 2 * HALO * C2;
        k_zero_halo<<<(n6 + 255) / 256, 256, 0, stream>>>(h6p, B, LP6, C2, HALO);
    }

    k_bnprep<<<1, C,  0, stream>>>(bn1p[0], bn1p[1], bn1p[2], bn1p[3], bn1i, bn1b, C);
    k_bnprep<<<1, C,  0, stream>>>(bn2p[0], bn2p[1], bn2p[2], bn2p[3], bn2i, bn2b, C);
    k_bnprep<<<1, C2, 0, stream>>>(bn3p[0], bn3p[1], bn3p[2], bn3p[3], bn3i, bn3b, C2);
    k_bnprep<<<1, C2, 0, stream>>>(bn5p[0], bn5p[1], bn5p[2], bn5p[3], bn5i, bn5b, C2);
    k_bnprep<<<1, C2, 0, stream>>>(bnfp[0], bnfp[1], bnfp[2], bnfp[3], bnfi, bnfb, C2);

    run_conv<0>(xb, w1b,  h1, nullptr, bn1i, bn1b, B, 128,  L0,  C,  L1, L1,  4, 0, stream);
    run_conv<7>(h1, p1wb, h2, nullptr, nullptr, nullptr, B, 512, L1, C, L2, LP2, 4, 0, stream);
    run_conv<7>(h2, w2b,  h3, nullptr, bn2i, bn2b, B, 512,  LP2, C,  L3, L3,  1, 2, stream);
    run_conv<7>(h3, p2wb, h4, nullptr, nullptr, nullptr, B, 512, L3, C, L4, LP4, 4, 0, stream);
    run_conv<7>(h4, w3b,  h5, nullptr, bn3i, bn3b, B, 512,  LP4, C2, L5, L5,  1, 1, stream);
    run_conv<8>(h5, p3wb, h6, nullptr, nullptr, nullptr, B, 1024, L5, C2, L6, LP6, 4, 0, stream);
    run_conv<8>(h6, w5b,  h7, nullptr, bn5i, bn5b, B, 768,  LP6, C2, L7, L7,  1, 2, stream);
    run_conv<8>(h7, p5wb, h8, nullptr, nullptr, nullptr, B, 1024, L7, C2, L8, L8, 4, 0, stream);
    run_conv<8>(h8, embwb, nullptr, h9f, nullptr, nullptr, B, 4096, L8, C2, 1, 1, 1, 0, stream);

    k_fc1<<<(B * C2 + 255) / 256, 256, 0, stream>>>(h9f, fc1w, bnfi, bnfb, h10, B, C2);
    k_fc2<<<(B * NC + 255) / 256, 256, 0, stream>>>(h10, fc2w, out, B, C2, NC);
}
